// MeanFieldIsing_75127567942221
// MI455X (gfx1250) — matrix operand order repaired and checked by reading, not compiled, not run on hardware
//
#include <hip/hip_runtime.h>
#include <hip/hip_fp16.h>

typedef __attribute__((ext_vector_type(16))) _Float16 v16h;
typedef __attribute__((ext_vector_type(2)))  _Float16 h2;
typedef __attribute__((ext_vector_type(8)))  float    v8f;
typedef __attribute__((ext_vector_type(4)))  unsigned u32x4;
typedef __attribute__((ext_vector_type(4)))  int      i32x4;
typedef __attribute__((ext_vector_type(8)))  int      i32x8;

#define NS 128
#define T_MF 25

__device__ __forceinline__ float fast_tanh(float x) {
#if defined(__HIP_DEVICE_COMPILE__) && __has_builtin(__builtin_amdgcn_tanhf)
  return __builtin_amdgcn_tanhf(x);
#else
  return tanhf(x);
#endif
}

// Cross-half swap (lane ^ 16) as a pure VALU op: v_permlanex16_b32 with
// identity lane selects. Falls back to __shfl_xor (ds_bpermute) if absent.
__device__ __forceinline__ unsigned swap_half(unsigned x) {
#if defined(__HIP_DEVICE_COMPILE__) && __has_builtin(__builtin_amdgcn_permlanex16)
  return (unsigned)__builtin_amdgcn_permlanex16(
      (int)x, (int)x, 0x76543210, 0xfedcba98, false, false);
#else
  return (unsigned)__shfl_xor((int)x, 16, 32);
#endif
}

// ---------------------------------------------------------------------------
// Prep: pack Jm^T (diag masked) into WMMA A-fragment order (f16) in d_ws.
// Fragment (mt,kt) = 16x32 tile of J^T, 512 halves; lane p reads its 32 bytes
// at frag*1024 + p*32. Half h of lane p (contiguous K order, matched by the B fragment build):
// row M = p%16, K = 16*(p/16) + h.
// ---------------------------------------------------------------------------
__global__ __launch_bounds__(256) void ising_prep_frag(
    const float* __restrict__ J, _Float16* __restrict__ frag) {
  int idx = blockIdx.x * 256 + threadIdx.x;     // 0..16383
  int fragId = idx >> 9;                        // mt*4 + kt
  int mt = fragId >> 2;
  int kt = fragId & 3;
  int within = idx & 511;
  int p = within >> 4;                          // "lane" 0..31
  int h = within & 15;                          // half 0..15
  int v = h >> 1, j = h & 1;
  int g = p >> 4, row = p & 15;
  int k_local = 16 * g + h;
  int m = 16 * mt + row;                        // output-spin (row of J^T)
  int k = 32 * kt + k_local;                    // input-spin  (col of J^T)
  float val = (k == m) ? 0.0f : J[k * NS + m];  // J^T[m][k] = J[k][m], masked
  frag[idx] = (_Float16)val;
}

// ---------------------------------------------------------------------------
// Main kernel: 256 threads = 8 waves; each wave owns 16 batch rows.
// State: s[mt][r] = spins[m = 16*mt + 8*g + r][batch col n = lane%16] (f32).
// Per iteration: c = lf ; c += J^T(frags) x spins^T via 32 WMMAs;
// spins = 0.7*tanh(4c) + 0.3*spins.
// ---------------------------------------------------------------------------
__global__ __launch_bounds__(256) void ising_mf_kernel(
    const float* __restrict__ features,
    const float* __restrict__ local_field,
    const _Float16* __restrict__ fragG,
    float* __restrict__ out_spins,
    float* __restrict__ out_fe) {
  __shared__ __align__(32) _Float16 lds_frag[16384];   // 32 KB A-fragments
  __shared__ __align__(32) float    lds_lf[NS];

  const int tid = threadIdx.x;

#if defined(__HIP_DEVICE_COMPILE__) && __has_builtin(__builtin_amdgcn_tensor_load_to_lds)
  // Tensor Data Mover: one DMA of the whole 32 KB fragment table into LDS.
  // D# describes a 1-D tensor of 16384 x 2-byte elements, tile == tensor.
  if (tid < 32) {  // wave 0 issues; TENSORcnt is per-wave
    unsigned lds_off = (unsigned)(size_t)(void*)lds_frag;   // LDS offset = low 32 bits
    unsigned long long ga = (unsigned long long)(size_t)fragG;
    u32x4 g0;
    g0[0] = 1u;                                             // count=1, user mode
    g0[1] = lds_off;                                        // lds_addr
    g0[2] = (unsigned)ga;                                   // global_addr[31:0]
    g0[3] = ((unsigned)(ga >> 32) & 0x1FFFFFFu) | (2u << 30); // addr[56:32] | type=2
    i32x8 g1;
    g1[0] = (int)(1u << 16);        // workgroup_mask=0, data_size=2B, no flags
    g1[1] = (int)(16384u << 16);    // tensor_dim0[15:0]=16384 (bits 63:48)
    g1[2] = (int)(1u << 16);        // tensor_dim0[31:16]=0 | tensor_dim1=1
    g1[3] = (int)(16384u << 16);    // tensor_dim1 hi=0 | tile_dim0=16384
    g1[4] = 1;                      // tile_dim1=1, tile_dim2=0
    g1[5] = 16384;                  // tensor_dim0_stride[31:0]
    g1[6] = 0;                      // stride hi | tensor_dim1_stride lo
    g1[7] = 0;
    i32x4 z4 = {0, 0, 0, 0};
#if __clang_major__ >= 23
    i32x8 z8 = {0, 0, 0, 0, 0, 0, 0, 0};
    __builtin_amdgcn_tensor_load_to_lds(g0, g1, z4, z4, z8, 0);
#else
    __builtin_amdgcn_tensor_load_to_lds(g0, g1, z4, z4, 0);
#endif
#if __has_builtin(__builtin_amdgcn_s_wait_tensorcnt)
    __builtin_amdgcn_s_wait_tensorcnt(0);
#else
    asm volatile("s_wait_tensorcnt 0x0" ::: "memory");
#endif
  }
#else
  {
    const float4* src = (const float4*)fragG;
    float4*       dst = (float4*)lds_frag;
#pragma unroll
    for (int i = 0; i < 8; ++i) dst[tid + 256 * i] = src[tid + 256 * i];
  }
#endif
  if (tid < NS) lds_lf[tid] = local_field[tid];
  __syncthreads();

  const int lane = tid & 31;
  const int wv   = tid >> 5;
  const int g    = lane >> 4;    // half-wave id
  const int n    = lane & 15;    // batch column within wave tile
  const bool g0  = (g == 0);
  const long batch = ((long)blockIdx.x * 8 + wv) * 16 + n;
  const float* frow = features + batch * NS;

  // spins0 = tanh(0.1 * features), loaded straight into D-layout
  float s[8][8];
#pragma unroll
  for (int mt = 0; mt < 8; ++mt) {
    float4 f0 = *(const float4*)(frow + 16 * mt + 8 * g);
    float4 f1 = *(const float4*)(frow + 16 * mt + 8 * g + 4);
    s[mt][0] = fast_tanh(0.1f * f0.x);
    s[mt][1] = fast_tanh(0.1f * f0.y);
    s[mt][2] = fast_tanh(0.1f * f0.z);
    s[mt][3] = fast_tanh(0.1f * f0.w);
    s[mt][4] = fast_tanh(0.1f * f1.x);
    s[mt][5] = fast_tanh(0.1f * f1.y);
    s[mt][6] = fast_tanh(0.1f * f1.z);
    s[mt][7] = fast_tanh(0.1f * f1.w);
  }

  v8f c[8];

  auto matmul = [&]() {
    // C init = local_field broadcast (lanes with same g hit same LDS words)
#pragma unroll
    for (int mt = 0; mt < 8; ++mt)
      c[mt] = *(const v8f*)(lds_lf + 16 * mt + 8 * g);
#pragma unroll
    for (int kt = 0; kt < 4; ++kt) {
      // Build B fragment (K=32 chunk of spins^T): pack to f16x2 first, then
      // a single VALU cross-half swap per dword.
      // lane g==0: dwords 0..3 = own s[2kt] pairs, 4..7 = partner s[2kt]
      // lane g==1: dwords 0..3 = partner s[2kt+1], 4..7 = own s[2kt+1]
      union { unsigned u[8]; v16h h; } bu;
#pragma unroll
      for (int d = 0; d < 4; ++d) {
        h2 pa, pb;
        pa[0] = (_Float16)s[2 * kt][2 * d];
        pa[1] = (_Float16)s[2 * kt][2 * d + 1];
        pb[0] = (_Float16)s[2 * kt + 1][2 * d];
        pb[1] = (_Float16)s[2 * kt + 1][2 * d + 1];
        unsigned ua = __builtin_bit_cast(unsigned, pa);
        unsigned ub = __builtin_bit_cast(unsigned, pb);
        unsigned own = g0 ? ua : ub;
        unsigned snd = g0 ? ub : ua;
        unsigned rcv = swap_half(snd);
        bu.u[d]     = g0 ? own : rcv;
        bu.u[4 + d] = g0 ? rcv : own;
      }
#pragma unroll
      for (int mt = 0; mt < 8; ++mt) {
        const v16h* ap =
            (const v16h*)(lds_frag + (mt * 4 + kt) * 512 + lane * 16);
        c[mt] = __builtin_amdgcn_wmma_f32_16x16x32_f16(
            false, *ap, false, bu.h, (short)0, c[mt], false, false);
      }
    }
  };

#pragma unroll 1
  for (int t = 0; t < T_MF; ++t) {
    matmul();
#pragma unroll
    for (int mt = 0; mt < 8; ++mt) {
#pragma unroll
      for (int r = 0; r < 8; ++r) {
        float nw = fast_tanh(4.0f * c[mt][r]);       // tanh(h_eff / 0.25)
        s[mt][r] = 0.7f * nw + 0.3f * s[mt][r];      // damped update
      }
    }
  }

  matmul();  // h_final with the converged spins

  // free_energy = -0.25 * sum_m log(2*cosh(4*h_final[m]))
  //             = -0.25 * sum_m (|x| + log(1 + exp(-2|x|))),  x = 4*h
  float partial = 0.0f;
#pragma unroll
  for (int mt = 0; mt < 8; ++mt) {
#pragma unroll
    for (int r = 0; r < 8; ++r) {
      float x  = 4.0f * c[mt][r];
      float ax = fabsf(x);
      partial += ax + __logf(1.0f + __expf(-2.0f * ax));
    }
  }
  float total =
      partial + __builtin_bit_cast(float, (unsigned)swap_half(
                    __builtin_bit_cast(unsigned, partial)));

  // write spins [B,S]
  float* orow = out_spins + batch * NS;
#pragma unroll
  for (int mt = 0; mt < 8; ++mt) {
    float4 a0 = make_float4(s[mt][0], s[mt][1], s[mt][2], s[mt][3]);
    float4 a1 = make_float4(s[mt][4], s[mt][5], s[mt][6], s[mt][7]);
    *(float4*)(orow + 16 * mt + 8 * g)     = a0;
    *(float4*)(orow + 16 * mt + 8 * g + 4) = a1;
  }
  if (g0) out_fe[batch] = -0.25f * total;
}

extern "C" void kernel_launch(void* const* d_in, const int* in_sizes, int n_in,
                              void* d_out, int out_size, void* d_ws, size_t ws_size,
                              hipStream_t stream) {
  const float* features = (const float*)d_in[0];
  const float* lf       = (const float*)d_in[1];
  const float* J        = (const float*)d_in[2];
  float* out = (float*)d_out;
  const long B = in_sizes[0] / NS;

  _Float16* frag = (_Float16*)d_ws;  // 32 KB of packed A-fragments
  ising_prep_frag<<<64, 256, 0, stream>>>(J, frag);

  const int blocks = (int)(B / 128);  // 8 waves x 16 rows per block
  ising_mf_kernel<<<blocks, 256, 0, stream>>>(
      features, lf, frag, out, out + (size_t)B * NS);
}
